// TransMIL_78855599554702
// MI455X (gfx1250) — matrix operand order repaired and checked by reading, not compiled, not run on hardware
//
#include <hip/hip_runtime.h>
#include <stdint.h>

// ---------------------------------------------------------------------------
// TransMIL for MI455X (gfx1250, wave32, WMMA).
// All matmuls run through one batched/strided WMMA GEMM:
//   - f16 operands, f32 accumulate via v_wmma_f32_16x16x32_f16
//   - B tiles staged into LDS with double-buffered GLOBAL_LOAD_ASYNC_TO_LDS_B128
//     (ASYNCcnt + workgroup barrier synchronization, CDNA5 async path)
//   - each wave computes a 32x64 strip (8 WMMA tiles), block = 8 waves = 256 rows
//   - edge strips are CLAMPED (m0 = min(m0, M-32)) instead of guarded: duplicate
//     waves recompute bit-identical values, so there is no divergence, no
//     exec-masked phi copies, and no accumulator spills in the inner loop
//   - fused alpha/bias/relu/f16-cast epilogues so activations make one trip
// ---------------------------------------------------------------------------

typedef _Float16 f16;
typedef __attribute__((ext_vector_type(16))) _Float16 v16h;
typedef __attribute__((ext_vector_type(8)))  _Float16 v8h;
typedef __attribute__((ext_vector_type(8)))  float    v8f;

#define B_      2
#define SEQ     10000
#define INDIM   1024
#define DMODEL  512
#define NHEADS  8
#define DH_     64
#define NTOK    10001      // SEQ + cls
#define NP_     10240      // padded to multiple of 256
#define PADN_   239        // zeros prepended
#define LFACT_  40         // NP_/256 landmark pooling factor
#define MLM     256
#define RESK    33
#define SCALE_  0.125f     // dh^-0.5

#define LDB_T   40         // BT LDS row stride (32 data + 8 pad halves)
#define LDB_N   72         // non-BT LDS row stride (64 data + 8 pad halves)

// ----------------------------- reductions ----------------------------------
__device__ __forceinline__ float blockReduceSum(float v, float* red) {
  int t = threadIdx.x;
  __syncthreads();
  red[t] = v; __syncthreads();
  for (int s = 128; s > 0; s >>= 1) { if (t < s) red[t] += red[t + s]; __syncthreads(); }
  float r = red[0]; __syncthreads();
  return r;
}
__device__ __forceinline__ float blockReduceMax(float v, float* red) {
  int t = threadIdx.x;
  __syncthreads();
  red[t] = v; __syncthreads();
  for (int s = 128; s > 0; s >>= 1) { if (t < s) red[t] = fmaxf(red[t], red[t + s]); __syncthreads(); }
  float r = red[0]; __syncthreads();
  return r;
}

// --------------------- async B-tile stage (global -> LDS) -------------------
// One b128 async load per thread per stage: 256 threads x 16B = 4KB tile.
//   BT  : tile is 64 rows (N) x 32 halves (K), LDS row stride LDB_T
//   !BT : tile is 32 rows (K) x 64 halves (N), LDS row stride LDB_N
// All address terms are 16B aligned at every call site (ldb in
// {64,256,512,1024,1536}, k0 % 32 == 0, n0 % 64 == 0, 256B-aligned arena).
template <bool BT>
__device__ __forceinline__ void stageB(const f16* __restrict__ Bm, int ldb,
                                       int n0, int k0, f16* dst, int t) {
  const f16* g;
  f16* l;
  if (BT) {
    const int row = t >> 2, ch = t & 3;              // 64 rows x 4 chunks
    g = Bm + (size_t)(n0 + row) * ldb + k0 + ch * 8;
    l = dst + row * LDB_T + ch * 8;
  } else {
    const int row = t >> 3, ch = t & 7;              // 32 rows x 8 chunks
    g = Bm + (size_t)(k0 + row) * ldb + n0 + ch * 8;
    l = dst + row * LDB_N + ch * 8;
  }
  const unsigned loff = (unsigned)(uintptr_t)l;       // LDS byte offset (addr[31:0])
  asm volatile("global_load_async_to_lds_b128 %0, %1, off"
               :: "v"(loff), "v"((unsigned long long)(uintptr_t)g)
               : "memory");
}

// ------------------------------- GEMM ---------------------------------------
// C[z] = act(alpha * A[z](MxK) x op(B[z]) + bias)
//   BT=true : B stored [N,K] row-major (weights) -> computes A @ B^T
//   BT=false: B stored [K,N] row-major           -> computes A @ B
// Batched via grid.z; per-z offsets decomposed as z = zb*Hdiv + zh so the
// head-interleaved qkv layout needs no transposes.
// grid.y covers ceil(M/256); M >= 32, M%16==0, N%64==0, K%32==0 at every call.
template <bool BT, bool OUT16>
__global__ void gemm_wmma(const f16* __restrict__ A, const f16* __restrict__ Bm,
                          void* __restrict__ Cv, const float* __restrict__ bias,
                          int M, int N, int K, int lda, int ldb, int ldc,
                          long sAb, long sAh, long sBb, long sBh, long sCb, long sCh,
                          int Hdiv, float alpha, int relu) {
  __shared__ f16 Bs[2][64 * LDB_T];   // 10KB: covers both tile layouts

  const int z  = blockIdx.z;
  const int zb = z / Hdiv, zh = z % Hdiv;
  A  += (size_t)zb * sAb + (size_t)zh * sAh;
  Bm += (size_t)zb * sBb + (size_t)zh * sBh;

  const int t    = threadIdx.x;
  const int wave = t >> 5;
  const int lane = t & 31;
  int m0 = (blockIdx.y * 8 + wave) * 32;   // wave's 32-row strip
  if (m0 > M - 32) m0 = M - 32;            // clamp edge strips (see header note)
  const int n0   = blockIdx.x * 64;
  const int half = lane >> 4;
  const int l15  = lane & 15;

  const f16* Arow0 = A + (size_t)(m0 + l15) * lda;
  const f16* Arow1 = Arow0 + (size_t)16 * lda;

  v8f zero = {0.f, 0.f, 0.f, 0.f, 0.f, 0.f, 0.f, 0.f};
  v8f acc0[4] = {zero, zero, zero, zero};
  v8f acc1[4] = {zero, zero, zero, zero};

  const int steps = K >> 5;
  stageB<BT>(Bm, ldb, n0, 0, &Bs[0][0], t);          // prologue stage 0

  for (int s = 0; s < steps; ++s) {
    const f16* Bcur = &Bs[s & 1][0];
    if (s + 1 < steps) {
      stageB<BT>(Bm, ldb, n0, (s + 1) * 32, &Bs[(s + 1) & 1][0], t);
      asm volatile("s_wait_asynccnt 0x1" ::: "memory");  // stage s landed
    } else {
      asm volatile("s_wait_asynccnt 0x0" ::: "memory");
    }
    __syncthreads();   // stage s visible to all waves; guards buffer reuse

    const int k0 = s * 32;
    // A fragments: contiguous K order matching B: VGPR j holds K pair
    // kk = 2j + 16*half; two b128 loads per fragment.
    v16h a0, a1;
#pragma unroll
    for (int j = 0; j < 8; ++j) {
      const int kk = 2 * j + 16 * half;
      union { unsigned u; f16 h[2]; } w0, w1;
      w0.u = *(const unsigned*)(Arow0 + k0 + kk);
      w1.u = *(const unsigned*)(Arow1 + k0 + kk);
      a0[2 * j] = w0.h[0]; a0[2 * j + 1] = w0.h[1];
      a1[2 * j] = w1.h[0]; a1[2 * j + 1] = w1.h[1];
    }
    if (k0 + 64 < K) __builtin_prefetch(Arow0 + k0 + 64, 0, 1);

#pragma unroll
    for (int t4 = 0; t4 < 4; ++t4) {
      const int nn = t4 * 16 + l15;     // column within the 64-wide tile
      v16h b;
      if (BT) {
        const v8h* p = (const v8h*)(Bcur + nn * LDB_T + 16 * half);
        v8h lo = p[0], hi = p[1];
#pragma unroll
        for (int e = 0; e < 8; ++e) { b[e] = lo[e]; b[8 + e] = hi[e]; }
      } else {
#pragma unroll
        for (int e = 0; e < 16; ++e)
          b[e] = Bcur[(16 * half + e) * LDB_N + nn];
      }
      acc0[t4] = __builtin_amdgcn_wmma_f32_16x16x32_f16(
          false, a0, false, b, (short)0, acc0[t4], false, false);
      acc1[t4] = __builtin_amdgcn_wmma_f32_16x16x32_f16(
          false, a1, false, b, (short)0, acc1[t4], false, false);
    }
  }

  // epilogue: alpha, bias, relu, optional f16 cast
  const long coff = (long)zb * sCb + (long)zh * sCh;
  float* Cf = (float*)Cv + coff;
  f16*   Ch = (f16*)Cv + coff;
#pragma unroll
  for (int t4 = 0; t4 < 4; ++t4) {
    const int col = n0 + t4 * 16 + l15;
    const float bv = bias ? bias[col] : 0.f;
#pragma unroll
    for (int r = 0; r < 8; ++r) {
      const int row0 = m0 + 8 * half + r;
      float v0 = acc0[t4][r] * alpha + bv;
      float v1 = acc1[t4][r] * alpha + bv;
      if (relu) { v0 = fmaxf(v0, 0.f); v1 = fmaxf(v1, 0.f); }
      if (OUT16) {
        Ch[(size_t)row0 * ldc + col] = (f16)v0;
        Ch[(size_t)(row0 + 16) * ldc + col] = (f16)v1;
      } else {
        Cf[(size_t)row0 * ldc + col] = v0;
        Cf[(size_t)(row0 + 16) * ldc + col] = v1;
      }
    }
  }
}

// --------------------------- elementwise ------------------------------------
__global__ void cast_f32_f16(const float* __restrict__ in, f16* __restrict__ o, long n) {
  long i = (long)blockIdx.x * 256 + threadIdx.x;
  if (i < n) o[i] = (f16)in[i];
}

// LayerNorm rows of x (B,NTOK,512) -> f16 padded buffer (B,NP,512) with
// PADN_ leading zero rows per batch (feeds the qkv GEMM directly).
__global__ void ln_to_padded(const float* __restrict__ x, const float* __restrict__ g,
                             const float* __restrict__ bta, f16* __restrict__ xp) {
  __shared__ float red[256];
  const int row = blockIdx.x;              // 0 .. B_*NP_-1
  const int bb = row / NP_, r = row % NP_;
  f16* out = xp + (size_t)row * DMODEL;
  const int t = threadIdx.x;
  if (r < PADN_) { out[t] = (f16)0.f; out[t + 256] = (f16)0.f; return; }
  const float* xr = x + ((size_t)bb * NTOK + (r - PADN_)) * DMODEL;
  const float a = xr[t], b2 = xr[t + 256];
  const float mu  = blockReduceSum(a + b2, red) * (1.f / 512.f);
  const float d0 = a - mu, d1 = b2 - mu;
  const float var = blockReduceSum(d0 * d0 + d1 * d1, red) * (1.f / 512.f);
  const float inv = rsqrtf(var + 1e-5f);
  out[t]       = (f16)(d0 * inv * g[t] + bta[t]);
  out[t + 256] = (f16)(d1 * inv * g[t + 256] + bta[t + 256]);
}

// Landmark means: q_l/k_l[z,m,dh] = mean over 40 contiguous rows (unscaled;
// the dh^-0.5 is folded into the sim GEMMs' alpha).
__global__ void landmarks(const f16* __restrict__ qkv, f16* __restrict__ ql,
                          f16* __restrict__ kl) {
  const int idx = blockIdx.x * 256 + threadIdx.x;  // 16*256*64 threads
  const int dh = idx & 63;
  const int m  = (idx >> 6) & 255;
  const int z  = idx >> 14;
  const int bb = z >> 3, h = z & 7;
  const f16* base = qkv + (size_t)bb * NP_ * 1536 + h * 64 + dh;
  const size_t rs = (size_t)(m * LFACT_) * 1536;
  float sq = 0.f, sk = 0.f;
  for (int j = 0; j < LFACT_; ++j) {
    const size_t off = rs + (size_t)j * 1536;
    sq += (float)base[off];
    sk += (float)base[off + 512];
  }
  ql[((size_t)z * 256 + m) * 64 + dh] = (f16)(sq * (1.f / LFACT_));
  kl[((size_t)z * 256 + m) * 64 + dh] = (f16)(sk * (1.f / LFACT_));
}

// In-place f16 row softmax (block per row).
__global__ void softmax_rows(f16* __restrict__ buf, int cols) {
  __shared__ float red[256];
  f16* row = buf + (size_t)blockIdx.x * cols;
  const int t = threadIdx.x;
  float mx = -3.0e38f;
  for (int c = t; c < cols; c += 256) mx = fmaxf(mx, (float)row[c]);
  mx = blockReduceMax(mx, red);
  float s = 0.f;
  for (int c = t; c < cols; c += 256) s += __expf((float)row[c] - mx);
  s = blockReduceSum(s, red);
  const float inv = 1.f / s;
  for (int c = t; c < cols; c += 256) row[c] = (f16)(__expf((float)row[c] - mx) * inv);
}

// z0 = A^T / (max(row_abs_sums) * max(col_abs_sums)) per 256x256 batch.
__global__ void pinv_init(const f16* __restrict__ a2, f16* __restrict__ z) {
  __shared__ float red[256];
  const f16* A = a2 + (size_t)blockIdx.x * 65536;
  f16* Z = z + (size_t)blockIdx.x * 65536;
  const int t = threadIdx.x;
  float rs = 0.f, cs = 0.f;
  for (int j = 0; j < 256; ++j) rs += fabsf((float)A[t * 256 + j]);
  for (int i = 0; i < 256; ++i) cs += fabsf((float)A[i * 256 + t]);
  const float maxcol = blockReduceMax(rs, red);
  const float maxrow = blockReduceMax(cs, red);
  const float inv = 1.f / (maxcol * maxrow);
  for (int i = 0; i < 256; ++i) Z[t * 256 + i] = (f16)((float)A[i * 256 + t] * inv);
}

// w = alpha*I - s   (16 batched 256x256 f16)
__global__ void axpy_ident(const f16* __restrict__ s, f16* __restrict__ w, float alpha) {
  const int i = blockIdx.x * 256 + threadIdx.x;    // 16*65536 total
  const int within = i & 65535;
  const int r = within >> 8, c = within & 255;
  w[i] = (f16)((r == c ? alpha : 0.f) - (float)s[i]);
}

// o(b,n,h*64+dh) += depthwise conv1d(v, res_w) along the padded sequence.
__global__ void dwconv1d_add(const f16* __restrict__ qkv, const float* __restrict__ wres,
                             float* __restrict__ o) {
  const size_t i = (size_t)blockIdx.x * 256 + threadIdx.x;
  if (i >= (size_t)B_ * NP_ * DMODEL) return;
  const int c  = (int)(i % DMODEL);
  const int n  = (int)((i / DMODEL) % NP_);
  const int bb = (int)(i / ((size_t)DMODEL * NP_));
  const int h = c >> 6, dh = c & 63;
  const f16* v = qkv + (size_t)bb * NP_ * 1536 + 1024 + h * 64 + dh;
  const float* w = wres + h * RESK;
  float s = 0.f;
#pragma unroll
  for (int tk = 0; tk < RESK; ++tk) {
    const int nn = n + tk - RESK / 2;
    if (nn >= 0 && nn < NP_) s += (float)v[(size_t)nn * 1536] * w[tk];
  }
  o[i] += s;
}

// x += proj[:, -NTOK:]  (rows PADN_.. of the padded projection)
__global__ void residual_add(const float* __restrict__ proj, float* __restrict__ x) {
  const size_t i = (size_t)blockIdx.x * 256 + threadIdx.x;
  if (i >= (size_t)B_ * NTOK * DMODEL) return;
  const int c = (int)(i % DMODEL);
  const size_t rr = i / DMODEL;
  const int r = (int)(rr % NTOK);
  const int bb = (int)(rr / NTOK);
  x[i] += proj[((size_t)bb * NP_ + PADN_ + r) * DMODEL + c];
}

// PPEG: y = fm + conv7 + b7 + conv5 + b5 + conv3 + b3 (fused, 100x100 map).
__global__ void ppeg_fused(const float* __restrict__ x,
                           const float* __restrict__ w7, const float* __restrict__ b7,
                           const float* __restrict__ w5, const float* __restrict__ b5,
                           const float* __restrict__ w3, const float* __restrict__ b3,
                           float* __restrict__ y) {
  const size_t i = (size_t)blockIdx.x * 256 + threadIdx.x;
  if (i >= (size_t)B_ * DMODEL * SEQ) return;
  const int p  = (int)(i % SEQ);
  const int c  = (int)((i / SEQ) % DMODEL);
  const int bb = (int)(i / ((size_t)SEQ * DMODEL));
  const int r = p / 100, cc = p % 100;
  const float* xb = x + ((size_t)bb * NTOK + 1) * DMODEL + c;
  float acc = xb[(size_t)p * DMODEL] + b7[c] + b5[c] + b3[c];
  const float* W7 = w7 + c * 49;
#pragma unroll
  for (int dy = -3; dy <= 3; ++dy)
#pragma unroll
    for (int dx = -3; dx <= 3; ++dx) {
      const int rr = r + dy, c2 = cc + dx;
      if (rr < 0 || rr >= 100 || c2 < 0 || c2 >= 100) continue;
      acc += xb[(size_t)(rr * 100 + c2) * DMODEL] * W7[(dy + 3) * 7 + (dx + 3)];
    }
  const float* W5 = w5 + c * 25;
#pragma unroll
  for (int dy = -2; dy <= 2; ++dy)
#pragma unroll
    for (int dx = -2; dx <= 2; ++dx) {
      const int rr = r + dy, c2 = cc + dx;
      if (rr < 0 || rr >= 100 || c2 < 0 || c2 >= 100) continue;
      acc += xb[(size_t)(rr * 100 + c2) * DMODEL] * W5[(dy + 2) * 5 + (dx + 2)];
    }
  const float* W3 = w3 + c * 9;
#pragma unroll
  for (int dy = -1; dy <= 1; ++dy)
#pragma unroll
    for (int dx = -1; dx <= 1; ++dx) {
      const int rr = r + dy, c2 = cc + dx;
      if (rr < 0 || rr >= 100 || c2 < 0 || c2 >= 100) continue;
      acc += xb[(size_t)(rr * 100 + c2) * DMODEL] * W3[(dy + 1) * 3 + (dx + 1)];
    }
  y[((size_t)bb * NTOK + 1 + p) * DMODEL + c] = acc;
}

__global__ void copy_cls_row(const float* __restrict__ x, float* __restrict__ y) {
  const int i = blockIdx.x * 256 + threadIdx.x;
  if (i >= B_ * DMODEL) return;
  const int bb = i / DMODEL, c = i % DMODEL;
  y[(size_t)bb * NTOK * DMODEL + c] = x[(size_t)bb * NTOK * DMODEL + c];
}

__global__ void set_cls_row(const float* __restrict__ cls, float* __restrict__ x) {
  const int i = blockIdx.x * 256 + threadIdx.x;
  if (i >= B_ * DMODEL) return;
  const int bb = i / DMODEL, c = i % DMODEL;
  x[(size_t)bb * NTOK * DMODEL + c] = cls[c];
}

__global__ void final_ln_cls(const float* __restrict__ x, const float* __restrict__ g,
                             const float* __restrict__ bta, float* __restrict__ out) {
  __shared__ float red[256];
  const int bb = blockIdx.x;
  const int t = threadIdx.x;
  const float* xr = x + (size_t)bb * NTOK * DMODEL;
  const float a = xr[t], b2 = xr[t + 256];
  const float mu  = blockReduceSum(a + b2, red) * (1.f / 512.f);
  const float d0 = a - mu, d1 = b2 - mu;
  const float var = blockReduceSum(d0 * d0 + d1 * d1, red) * (1.f / 512.f);
  const float inv = rsqrtf(var + 1e-5f);
  out[bb * DMODEL + t]       = d0 * inv * g[t] + bta[t];
  out[bb * DMODEL + t + 256] = d1 * inv * g[t + 256] + bta[t + 256];
}

// ------------------------------- host ---------------------------------------
extern "C" void kernel_launch(void* const* d_in, const int* in_sizes, int n_in,
                              void* d_out, int out_size, void* d_ws, size_t ws_size,
                              hipStream_t stream) {
  const float* h     = (const float*)d_in[0];
  const float* pe_w  = (const float*)d_in[1];
  const float* pe_b  = (const float*)d_in[2];
  const float* cls   = (const float*)d_in[3];
  const float* ln_g  = (const float*)d_in[4];
  const float* ln_b  = (const float*)d_in[5];
  const float* qkv_w = (const float*)d_in[6];
  const float* ao_w  = (const float*)d_in[7];
  const float* ao_b  = (const float*)d_in[8];
  const float* resw  = (const float*)d_in[9];
  const float* c7w   = (const float*)d_in[10];
  const float* c7b   = (const float*)d_in[11];
  const float* c5w   = (const float*)d_in[12];
  const float* c5b   = (const float*)d_in[13];
  const float* c3w   = (const float*)d_in[14];
  const float* c3b   = (const float*)d_in[15];
  const float* fn_g  = (const float*)d_in[16];
  const float* fn_b  = (const float*)d_in[17];
  float* out = (float*)d_out;

  // bump allocator in d_ws (~420 MB peak), 256B aligned
  char* base = (char*)d_ws;
  size_t off = 0;
  auto alloc = [&](size_t bytes) -> void* {
    off = (off + 255) & ~(size_t)255;
    void* p = base + off;
    off += bytes;
    return p;
  };
  f16* hh     = (f16*)alloc((size_t)B_ * SEQ * INDIM * 2);
  f16* pewh   = (f16*)alloc((size_t)DMODEL * INDIM * 2);
  f16* qkvwh  = (f16*)alloc((size_t)2 * 1536 * DMODEL * 2);
  f16* aowh   = (f16*)alloc((size_t)2 * DMODEL * DMODEL * 2);
  float* xA   = (float*)alloc((size_t)B_ * NTOK * DMODEL * 4);
  float* xB   = (float*)alloc((size_t)B_ * NTOK * DMODEL * 4);
  f16* xp     = (f16*)alloc((size_t)B_ * NP_ * DMODEL * 2);
  f16* qkvh   = (f16*)alloc((size_t)B_ * NP_ * 1536 * 2);
  f16* ql     = (f16*)alloc((size_t)16 * 256 * 64 * 2);
  f16* kl     = (f16*)alloc((size_t)16 * 256 * 64 * 2);
  f16* simbuf = (f16*)alloc((size_t)16 * NP_ * 256 * 2);   // reused: sim3 then sim1
  f16* a2m    = (f16*)alloc((size_t)16 * 65536 * 2);
  f16* zA     = (f16*)alloc((size_t)16 * 65536 * 2);
  f16* zB     = (f16*)alloc((size_t)16 * 65536 * 2);
  f16* xzb    = (f16*)alloc((size_t)16 * 65536 * 2);
  f16* wb     = (f16*)alloc((size_t)16 * 65536 * 2);
  f16* pv     = (f16*)alloc((size_t)16 * 65536 * 2);
  f16* a3v    = (f16*)alloc((size_t)16 * 256 * 64 * 2);
  f16* tmat   = (f16*)alloc((size_t)16 * 256 * 64 * 2);
  float* obuf = (float*)alloc((size_t)B_ * NP_ * DMODEL * 4);
  f16* obufh  = (f16*)alloc((size_t)B_ * NP_ * DMODEL * 2);
  float* proj = (float*)alloc((size_t)B_ * NP_ * DMODEL * 4);

  auto cast = [&](const float* s, f16* d, long n) {
    cast_f32_f16<<<(unsigned)((n + 255) / 256), 256, 0, stream>>>(s, d, n);
  };

  // --- weight/input casts (done every call; no persistent state allowed) ---
  cast(h, hh, (long)B_ * SEQ * INDIM);
  cast(pe_w, pewh, (long)DMODEL * INDIM);
  cast(qkv_w, qkvwh, (long)2 * 1536 * DMODEL);
  cast(ao_w, aowh, (long)2 * DMODEL * DMODEL);

  // --- x = relu(h @ pe_w^T + pe_b) into rows 1.. of xA (batched over B) ---
  gemm_wmma<true, false><<<dim3(DMODEL / 64, 40, B_), 256, 0, stream>>>(
      hh, pewh, xA + DMODEL /*row 1*/, pe_b,
      SEQ, DMODEL, INDIM, INDIM, INDIM, DMODEL,
      (long)SEQ * INDIM, 0, 0, 0, (long)NTOK * DMODEL, 0, 1, 1.f, 1);
  set_cls_row<<<4, 256, 0, stream>>>(cls, xA);

  const long sQH = (long)NP_ * 1536;      // qkv per-batch stride
  const long sSH = (long)NP_ * 256;       // sim per-(b,h) stride (2621440)

  for (int l = 0; l < 2; ++l) {
    float* xin  = l ? xB : xA;
    float* xout = l ? xA : xB;

    // LayerNorm -> zero-padded f16 activations
    ln_to_padded<<<B_ * NP_, 256, 0, stream>>>(xin, ln_g + l * DMODEL, ln_b + l * DMODEL, xp);

    // qkv = xp @ qkv_w[l]^T  (M=20480, N=1536, K=512), f16 out
    gemm_wmma<true, true><<<dim3(1536 / 64, 80, 1), 256, 0, stream>>>(
        xp, qkvwh + (size_t)l * 1536 * DMODEL, qkvh, nullptr,
        B_ * NP_, 1536, DMODEL, DMODEL, DMODEL, 1536,
        0, 0, 0, 0, 0, 0, 1, 1.f, 0);

    landmarks<<<1024, 256, 0, stream>>>(qkvh, ql, kl);

    // sim2 = SCALE * q_l @ k_l^T  -> a2 (16 x 256 x 256)
    gemm_wmma<true, true><<<dim3(4, 1, 16), 256, 0, stream>>>(
        ql, kl, a2m, nullptr, 256, 256, 64, 64, 64, 256,
        16384, 0, 16384, 0, 65536, 0, 1, SCALE_, 0);
    softmax_rows<<<16 * 256, 256, 0, stream>>>(a2m, 256);

    // Moore-Penrose pinv (6 Newton-Schulz iterations) on the 16 batches
    pinv_init<<<16, 256, 0, stream>>>(a2m, zA);
    f16 *zc = zA, *zn = zB;
    for (int it = 0; it < 6; ++it) {
      gemm_wmma<false, true><<<dim3(4, 1, 16), 256, 0, stream>>>(   // xz = a2 @ z
          a2m, zc, xzb, nullptr, 256, 256, 256, 256, 256, 256,
          65536, 0, 65536, 0, 65536, 0, 1, 1.f, 0);
      axpy_ident<<<4096, 256, 0, stream>>>(xzb, wb, 7.f);           // w = 7I - xz
      gemm_wmma<false, true><<<dim3(4, 1, 16), 256, 0, stream>>>(   // pv = xz @ w
          xzb, wb, pv, nullptr, 256, 256, 256, 256, 256, 256,
          65536, 0, 65536, 0, 65536, 0, 1, 1.f, 0);
      axpy_ident<<<4096, 256, 0, stream>>>(pv, wb, 15.f);           // w = 15I - pv
      gemm_wmma<false, true><<<dim3(4, 1, 16), 256, 0, stream>>>(   // pv = xz @ w
          xzb, wb, pv, nullptr, 256, 256, 256, 256, 256, 256,
          65536, 0, 65536, 0, 65536, 0, 1, 1.f, 0);
      axpy_ident<<<4096, 256, 0, stream>>>(pv, wb, 13.f);           // w = 13I - pv
      gemm_wmma<false, true><<<dim3(4, 1, 16), 256, 0, stream>>>(   // zn = 0.25 z @ w
          zc, wb, zn, nullptr, 256, 256, 256, 256, 256, 256,
          65536, 0, 65536, 0, 65536, 0, 1, 0.25f, 0);
      f16* tsw = zc; zc = zn; zn = tsw;
    }

    // sim3 = SCALE * q_l @ k^T  (16 x 256 x 10240) into simbuf, softmax, a3v
    gemm_wmma<true, true><<<dim3(160, 1, 16), 256, 0, stream>>>(
        ql, qkvh + 512, simbuf, nullptr, 256, NP_, 64, 64, 1536, NP_,
        8 * 16384, 16384, sQH, 64, 8 * sSH, sSH, 8, SCALE_, 0);
    softmax_rows<<<16 * 256, 256, 0, stream>>>(simbuf, NP_);
    gemm_wmma<false, true><<<dim3(1, 1, 16), 256, 0, stream>>>(     // a3v = a3 @ v
        simbuf, qkvh + 1024, a3v, nullptr, 256, 64, NP_, NP_, 1536, 64,
        8 * sSH, sSH, sQH, 64, 8 * 16384, 16384, 8, 1.f, 0);

    // sim1 = SCALE * q @ k_l^T (16 x 10240 x 256) reusing simbuf, softmax
    gemm_wmma<true, true><<<dim3(4, 40, 16), 256, 0, stream>>>(
        qkvh, kl, simbuf, nullptr, NP_, 256, 64, 1536, 64, 256,
        sQH, 64, 8 * 16384, 16384, 8 * sSH, sSH, 8, SCALE_, 0);
    softmax_rows<<<16 * NP_, 256, 0, stream>>>(simbuf, 256);

    // t = a2_inv @ a3v ; o = a1 @ t  (written head-concatenated: (b,n,h*64+dh))
    gemm_wmma<false, true><<<dim3(1, 1, 16), 256, 0, stream>>>(
        zc, a3v, tmat, nullptr, 256, 64, 256, 256, 64, 64,
        65536, 0, 16384, 0, 16384, 0, 1, 1.f, 0);
    gemm_wmma<false, false><<<dim3(1, 40, 16), 256, 0, stream>>>(
        simbuf, tmat, obuf, nullptr, NP_, 64, 256, 256, 64, DMODEL,
        8 * sSH, sSH, 8 * 16384, 16384, (long)NP_ * DMODEL, 64, 8, 1.f, 0);

    // + depthwise conv1d(v) residual, then f16, then output projection
    dwconv1d_add<<<40960, 256, 0, stream>>>(qkvh, resw + l * NHEADS * RESK, obuf);
    cast(obuf, obufh, (long)B_ * NP_ * DMODEL);
    gemm_wmma<true, false><<<dim3(DMODEL / 64, 80, 1), 256, 0, stream>>>(
        obufh, aowh + (size_t)l * DMODEL * DMODEL, proj, ao_b + l * DMODEL,
        B_ * NP_, DMODEL, DMODEL, DMODEL, DMODEL, DMODEL,
        0, 0, 0, 0, 0, 0, 1, 1.f, 0);
    residual_add<<<40005, 256, 0, stream>>>(proj, xin);

    // PPEG (fused 7/5/3 depthwise convs on the 100x100 map), cls passthrough
    ppeg_fused<<<40000, 256, 0, stream>>>(xin, c7w, c7b, c5w, c5b, c3w, c3b, xout);
    copy_cls_row<<<4, 256, 0, stream>>>(xin, xout);
  }

  // final LayerNorm on the cls token only -> (B, 512)
  final_ln_cls<<<B_, 256, 0, stream>>>(xA, fn_g, fn_b, out);
}
